// SimplifiedS4Layer_10900626997666
// MI455X (gfx1250) — matrix operand order repaired and checked by reading, not compiled, not run on hardware
//
#include <hip/hip_runtime.h>

typedef __bf16 bf16_t;
typedef __attribute__((ext_vector_type(8)))  bf16_t v8bf;
typedef __attribute__((ext_vector_type(16))) bf16_t v16bf;
typedef __attribute__((ext_vector_type(8)))  float  v8f;

constexpr int BATCH = 4;
constexpr int SEQ   = 4096;
constexpr int DM    = 1024;
constexpr int NS    = 64;            // state size
constexpr int LCH   = 64;            // chunk length
constexpr int GCH   = SEQ / LCH;     // 64 chunks
constexpr int COLS  = BATCH * GCH;   // 256 matmul columns per channel
constexpr int PERD  = BATCH * SEQ;   // 16384 elems of Xt per d (== NS*COLS)
constexpr int LSTR  = 72;            // padded LDS row stride (bf16 elems)

__device__ __forceinline__ float softplus_f(float v) {
    return __logf(1.0f + __expf(v));
}

// A-operand (16x32 bf16): lane half selects K {0..7,16..23} vs {8..15,24..31}
__device__ __forceinline__ v16bf load_a_tile(const bf16_t* base, int m, int k0, int half) {
    const int ka = k0 + half * 8;
    v8bf lo = *(const v8bf*)(base + m * LSTR + ka);
    v8bf hi = *(const v8bf*)(base + m * LSTR + ka + 16);
    return __builtin_shufflevector(lo, hi, 0,1,2,3,4,5,6,7,8,9,10,11,12,13,14,15);
}

// B-operand (32x16 bf16): lane half selects K {0..7,16..23} vs {8..15,24..31}, col fixed per lane
__device__ __forceinline__ v16bf load_b_tile(const bf16_t* base, int col, int k0, int half) {
    const int kb = k0 + half * 8;
    v8bf lo = *(const v8bf*)(base + col * LSTR + kb);
    v8bf hi = *(const v8bf*)(base + col * LSTR + kb + 16);
    return __builtin_shufflevector(lo, hi, 0,1,2,3,4,5,6,7,8,9,10,11,12,13,14,15);
}

// ---------------- Kernel 1: x (b,s,d) f32 -> Xt[d][b][s] bf16 ----------------
__global__ void k_transpose_in(const float* __restrict__ x, bf16_t* __restrict__ Xt) {
    __shared__ float tile[64][65];
    const int d0 = blockIdx.x * 64, s0 = blockIdx.y * 64, b = blockIdx.z;
    const int j = threadIdx.x & 63, i0 = threadIdx.x >> 6;
#pragma unroll
    for (int r = 0; r < 16; ++r) {
        int i = i0 + r * 4;   // s offset
        tile[i][j] = x[(size_t)(b * SEQ + s0 + i) * DM + d0 + j];
    }
    __syncthreads();
    const int sp = (threadIdx.x & 31) * 2, dd0 = threadIdx.x >> 5;
#pragma unroll
    for (int r = 0; r < 8; ++r) {
        int dd = dd0 + r * 8; // d offset
        unsigned short u0 = __builtin_bit_cast(unsigned short, (bf16_t)tile[sp][dd]);
        unsigned short u1 = __builtin_bit_cast(unsigned short, (bf16_t)tile[sp + 1][dd]);
        unsigned u = (unsigned)u0 | ((unsigned)u1 << 16);
        *(unsigned*)&Xt[(size_t)(d0 + dd) * PERD + b * SEQ + s0 + sp] = u;
    }
}

// ------- Kernel 2: V[d][n][col] = P_d (64n x 64s) x X (64s x 256col), WMMA -------
__global__ void k_chunk_summary(const bf16_t* __restrict__ Xt,
                                const float* __restrict__ A, const float* __restrict__ Bm,
                                const float* __restrict__ delta, float* __restrict__ V) {
    __shared__ bf16_t ldsX[COLS * LSTR]; // [col][s]
    __shared__ bf16_t ldsP[NS * LSTR];   // [n][s]
    const int d = blockIdx.x, tid = threadIdx.x;
    const float dt = softplus_f(delta[d]);

    const unsigned* src = (const unsigned*)(Xt + (size_t)d * PERD);
    for (int k = tid; k < PERD / 2; k += 256) {
        int g = k * 2, col = g >> 6, s = g & 63;
        *(unsigned*)&ldsX[col * LSTR + s] = src[k];
    }
    for (int idx = tid; idx < NS * LCH; idx += 256) {
        int n = idx >> 6, s = idx & 63;
        float e = dt * A[d * NS + n];
        float val = (dt * Bm[d * NS + n]) * __expf(e * (float)(63 - s));
        ldsP[n * LSTR + s] = (bf16_t)val;
    }
    __syncthreads();

    const int wave = tid >> 5, lane = tid & 31, half = lane >> 4, lr = lane & 15;
    const int ct0 = wave * 2, ct1 = wave * 2 + 1;
    float* vg = V + (size_t)d * NS * COLS;

    v8f acc[8]; // [mt][ct pair]
#pragma unroll
    for (int i = 0; i < 8; ++i) {
        v8f z = {0, 0, 0, 0, 0, 0, 0, 0};
        acc[i] = z;
    }
#pragma unroll
    for (int ks = 0; ks < 2; ++ks) {
        // B tiles depend only on (ct, ks): load once, reuse across all mt
        v16bf b0 = load_b_tile(ldsX, ct0 * 16 + lr, ks * 32, half);
        v16bf b1 = load_b_tile(ldsX, ct1 * 16 + lr, ks * 32, half);
#pragma unroll
        for (int mt = 0; mt < 4; ++mt) {
            v16bf a = load_a_tile(ldsP, mt * 16 + lr, ks * 32, half);
            acc[mt * 2 + 0] = __builtin_amdgcn_wmma_f32_16x16x32_bf16(
                false, a, false, b0, (short)0, acc[mt * 2 + 0], false, false);
            acc[mt * 2 + 1] = __builtin_amdgcn_wmma_f32_16x16x32_bf16(
                false, a, false, b1, (short)0, acc[mt * 2 + 1], false, false);
        }
    }
#pragma unroll
    for (int mt = 0; mt < 4; ++mt) {
#pragma unroll
        for (int i = 0; i < 8; ++i) {
            int n = mt * 16 + half * 8 + i;
            vg[n * COLS + ct0 * 16 + lr] = acc[mt * 2 + 0][i];
            vg[n * COLS + ct1 * 16 + lr] = acc[mt * 2 + 1][i];
        }
    }
}

// ------- Kernel 3: cross-chunk scan over c (in place, via LDS): V -> entry states -------
__global__ void k_chunk_scan(float* __restrict__ V, const float* __restrict__ A,
                             const float* __restrict__ delta) {
    __shared__ float lv[128 * 69];
    const int d = blockIdx.x >> 1, nh = blockIdx.x & 1, tid = threadIdx.x;
    float* vg = V + (size_t)d * NS * COLS + nh * 32 * COLS;
    for (int k = tid; k < 32 * COLS; k += 128) {
        int n = k >> 8, col = k & 255;
        lv[(n * 4 + (col >> 6)) * 69 + (col & 63)] = vg[k];
    }
    __syncthreads();
    const int n = tid >> 2;
    const float dt = softplus_f(delta[d]);
    const float aL = __expf(64.0f * dt * A[d * NS + nh * 32 + n]);
    float carry = 0.0f;
    float* row = &lv[tid * 69];
#pragma unroll 1
    for (int c = 0; c < 64; ++c) {
        float v = row[c];
        row[c] = carry;               // state at entry of chunk c
        carry = fmaf(aL, carry, v);   // state at entry of chunk c+1
    }
    __syncthreads();
    for (int k = tid; k < 32 * COLS; k += 128) {
        int n2 = k >> 8, col = k & 255;
        vg[k] = lv[(n2 * 4 + (col >> 6)) * 69 + (col & 63)];
    }
}

// ------- Kernel 4: Y = T_d * X + W_d * U  (WMMA), Y(bf16) overlays Xt region -------
__global__ void k_output(bf16_t* __restrict__ Xt, const float* __restrict__ V,
                         const float* __restrict__ A, const float* __restrict__ Bm,
                         const float* __restrict__ C, const float* __restrict__ Dv,
                         const float* __restrict__ delta) {
    __shared__ __align__(16) char smemXU[2 * 128 * LSTR * 2]; // X + U, later reused as f32 Y
    __shared__ bf16_t ldsT[NS * LSTR]; // [t][s] Toeplitz
    __shared__ bf16_t ldsW[NS * LSTR]; // [t][n]
    __shared__ float kk[64];
    bf16_t* ldsX = (bf16_t*)smemXU;              // [colr][s]
    bf16_t* ldsU = (bf16_t*)smemXU + 128 * LSTR; // [colr][n]
    const int d = blockIdx.x, ch = blockIdx.y, tid = threadIdx.x;
    const int colbase = ch * 128;
    const float dt = softplus_f(delta[d]);

    const unsigned* src = (const unsigned*)(Xt + (size_t)d * PERD + colbase * 64);
    for (int k = tid; k < 4096; k += 256) {
        int g = k * 2, colr = g >> 6, s = g & 63;
        *(unsigned*)&ldsX[colr * LSTR + s] = src[k];
    }
    const float* vg = V + (size_t)d * NS * COLS;
    for (int idx = tid; idx < 8192; idx += 256) {
        int n = idx >> 7, colr = idx & 127;
        ldsU[colr * LSTR + n] = (bf16_t)vg[n * COLS + colbase + colr];
    }
    for (int idx = tid; idx < 4096; idx += 256) {
        int t = idx >> 6, n = idx & 63;
        ldsW[t * LSTR + n] = (bf16_t)(C[d * NS + n] * __expf(dt * A[d * NS + n] * (float)(t + 1)));
    }
    if (tid < 64) {
        float s = (tid == 0) ? Dv[d] : 0.0f;
        for (int n = 0; n < NS; ++n)
            s += C[d * NS + n] * (dt * Bm[d * NS + n]) * __expf(dt * A[d * NS + n] * (float)tid);
        kk[tid] = s;
    }
    __syncthreads();
    for (int idx = tid; idx < 4096; idx += 256) {
        int t = idx >> 6, sc = idx & 63;
        ldsT[t * LSTR + sc] = (sc <= t) ? (bf16_t)kk[t - sc] : (bf16_t)0.0f;
    }
    __syncthreads();

    const int wave = tid >> 5, lane = tid & 31, half = lane >> 4, lr = lane & 15;
    const int ct = wave; // 8 waves <-> 8 column tiles of this 128-col half
    v8f acc[4];
#pragma unroll
    for (int mt = 0; mt < 4; ++mt) {
        v8f z = {0, 0, 0, 0, 0, 0, 0, 0};
        acc[mt] = z;
    }
#pragma unroll
    for (int ks = 0; ks < 2; ++ks) {
        // B tiles depend only on ks for this wave's fixed ct: hoist out of mt loop
        v16bf bX = load_b_tile(ldsX, ct * 16 + lr, ks * 32, half);
        v16bf bU = load_b_tile(ldsU, ct * 16 + lr, ks * 32, half);
#pragma unroll
        for (int mt = 0; mt < 4; ++mt) {
            v16bf aT = load_a_tile(ldsT, mt * 16 + lr, ks * 32, half);
            acc[mt] = __builtin_amdgcn_wmma_f32_16x16x32_bf16(
                false, aT, false, bX, (short)0, acc[mt], false, false);
            v16bf aW = load_a_tile(ldsW, mt * 16 + lr, ks * 32, half);
            acc[mt] = __builtin_amdgcn_wmma_f32_16x16x32_bf16(
                false, aW, false, bU, (short)0, acc[mt], false, false);
        }
    }
    __syncthreads(); // X/U consumed; reuse region as Y
    float* ldsY = (float*)smemXU; // [t][colr], stride 133 (64*133*4 = 34048 <= 36864)
#pragma unroll
    for (int mt = 0; mt < 4; ++mt)
#pragma unroll
        for (int i = 0; i < 8; ++i) {
            int t = mt * 16 + half * 8 + i;
            ldsY[t * 133 + ct * 16 + lr] = acc[mt][i];
        }
    __syncthreads();
    unsigned* dst = (unsigned*)(Xt + (size_t)d * PERD + colbase * 64); // overlay
    for (int k = tid; k < 4096; k += 256) {
        int t2 = (k & 31) * 2, colr = k >> 5;
        unsigned short u0 = __builtin_bit_cast(unsigned short, (bf16_t)ldsY[t2 * 133 + colr]);
        unsigned short u1 = __builtin_bit_cast(unsigned short, (bf16_t)ldsY[(t2 + 1) * 133 + colr]);
        dst[colr * 32 + (t2 >> 1)] = (unsigned)u0 | ((unsigned)u1 << 16);
    }
}

// ---------------- Kernel 5: Yt[d][col][t] bf16 -> out (b,s,d) f32 ----------------
__global__ void k_transpose_out(const bf16_t* __restrict__ Yt, float* __restrict__ out) {
    __shared__ float tile[32][33];
    const int d0 = blockIdx.x * 32, s0 = blockIdx.y * 32, b = blockIdx.z;
    const int c = s0 >> 6, t0 = s0 & 63;
    const int tid = threadIdx.x;
    {
        const int j2 = (tid & 15) * 2, i0 = tid >> 4;
#pragma unroll
        for (int r = 0; r < 2; ++r) {
            int i = i0 + r * 16; // d offset
            unsigned u = *(const unsigned*)&Yt[(size_t)(d0 + i) * PERD + (b * 64 + c) * 64 + t0 + j2];
            tile[i][j2]     = (float)__builtin_bit_cast(bf16_t, (unsigned short)(u & 0xffffu));
            tile[i][j2 + 1] = (float)__builtin_bit_cast(bf16_t, (unsigned short)(u >> 16));
        }
    }
    __syncthreads();
    {
        const int i2 = tid & 31, j0 = tid >> 5;
#pragma unroll
        for (int r = 0; r < 4; ++r) {
            int j = j0 + r * 8;
            out[(size_t)(b * SEQ + s0 + j) * DM + d0 + i2] = tile[i2][j];
        }
    }
}

extern "C" void kernel_launch(void* const* d_in, const int* in_sizes, int n_in,
                              void* d_out, int out_size, void* d_ws, size_t ws_size,
                              hipStream_t stream) {
    (void)in_sizes; (void)n_in; (void)out_size; (void)ws_size;
    const float* x     = (const float*)d_in[0];
    const float* A     = (const float*)d_in[1];
    const float* Bm    = (const float*)d_in[2];
    const float* C     = (const float*)d_in[3];
    const float* Dv    = (const float*)d_in[4];
    const float* delta = (const float*)d_in[5];
    float* out = (float*)d_out;

    bf16_t* Xt = (bf16_t*)d_ws;                                        // 32 MB
    float*  V  = (float*)((char*)d_ws + (size_t)DM * PERD * 2);        // 64 MB

    k_transpose_in <<<dim3(DM / 64, SEQ / 64, BATCH), 256, 0, stream>>>(x, Xt);
    k_chunk_summary<<<DM, 256, 0, stream>>>(Xt, A, Bm, delta, V);
    k_chunk_scan   <<<DM * 2, 128, 0, stream>>>(V, A, delta);
    k_output       <<<dim3(DM, 2), 256, 0, stream>>>(Xt, V, A, Bm, C, Dv, delta);
    k_transpose_out<<<dim3(DM / 32, SEQ / 32, BATCH), 256, 0, stream>>>(Xt, out);
}
